// MMEncoder_79405355368577
// MI455X (gfx1250) — compile-verified, matrix operand order checked by reading, not run on hardware
//
#include <hip/hip_runtime.h>

#define T_STEPS 256
#define BSZ 64

typedef __attribute__((ext_vector_type(16))) __bf16 v16bf;
typedef __attribute__((ext_vector_type(8)))  __bf16 v8bf;
typedef __attribute__((ext_vector_type(8)))  float  v8f;

__device__ __forceinline__ __bf16 f2bf(float f) {
  unsigned u = __builtin_bit_cast(unsigned, f);
  unsigned r = u + 0x7FFFu + ((u >> 16) & 1u);
  unsigned short h = (unsigned short)(r >> 16);
  return __builtin_bit_cast(__bf16, h);
}
__device__ __forceinline__ unsigned short f2bfu(float f) {
  return __builtin_bit_cast(unsigned short, f2bf(f));
}
__device__ __forceinline__ float bf2f(unsigned short h) {
  unsigned u = (unsigned)h << 16;
  return __builtin_bit_cast(float, u);
}

// fp32 -> bf16 bulk cast; each thread converts 8 elements (b128 in/out)
__global__ void cast_f32_bf16(const float* __restrict__ src,
                              unsigned short* __restrict__ dst, long n8) {
  long i = (long)blockIdx.x * blockDim.x + threadIdx.x;
  if (i >= n8) return;
  const float4* p = (const float4*)(src + i * 8);
  float4 a = p[0], b = p[1];
  v8bf o;
  o[0] = f2bf(a.x); o[1] = f2bf(a.y); o[2] = f2bf(a.z); o[3] = f2bf(a.w);
  o[4] = f2bf(b.x); o[5] = f2bf(b.y); o[6] = f2bf(b.z); o[7] = f2bf(b.w);
  *(v8bf*)(dst + i * 8) = o;
}

// Load one 16x32 bf16 WMMA A/B fragment row for this lane from bf16 memory.
// Lanes 0-15 hold K = k0+0..7 and k0+16..23; lanes 16-31 hold K = k0+8..15
// and k0+24..31 -> two contiguous 8-element (16B) runs -> 2x global_load_b128.
__device__ __forceinline__ v16bf load_frag(const unsigned short* __restrict__ row,
                                           int k0, int half) {
  v8bf lo = *(const v8bf*)(row + k0 + half * 8);
  v8bf hi = *(const v8bf*)(row + k0 + 16 + half * 8);
  return __builtin_shufflevector(lo, hi, 0, 1, 2, 3, 4, 5, 6, 7,
                                 8, 9, 10, 11, 12, 13, 14, 15);
}

// Y[M,N] = A[M,K] @ W[N,K]^T + b0[N] + b1[N]
// A, W bf16; Y -> Yf (fp32) or Yh (bf16), the other null.
// Each wave computes a 16x64 strip: 1 A-frag reused over 4 B-frags / 4 wmma.
// Requires N % 64 == 0, M % 16 == 0, K % 32 == 0.
__global__ __launch_bounds__(256) void wmma_gemm_bias(
    const unsigned short* __restrict__ A, const unsigned short* __restrict__ W,
    const float* __restrict__ b0, const float* __restrict__ b1,
    float* __restrict__ Yf, unsigned short* __restrict__ Yh,
    int M, int N, int K) {
  int lane = threadIdx.x & 31;
  int wave = threadIdx.x >> 5;
  int half = lane >> 4;
  int idx  = lane & 15;
  int ngrp = N >> 6;
  long grp = (long)blockIdx.x * 8 + wave;
  long tot = (long)(M >> 4) * ngrp;
  if (grp >= tot) return;                    // wave-uniform: EXEC stays full
  int mt = (int)(grp / ngrp);
  int ng = (int)(grp % ngrp);
  int m0 = mt << 4;
  int n0 = ng << 6;
  const unsigned short* arow = A + (long)(m0 + idx) * K;
  const unsigned short* brow[4];
#pragma unroll
  for (int t = 0; t < 4; ++t) brow[t] = W + (long)(n0 + t * 16 + idx) * K;
  v8f acc[4] = {{}, {}, {}, {}};
  for (int k0 = 0; k0 < K; k0 += 32) {
    v16bf af = load_frag(arow, k0, half);
#pragma unroll
    for (int t = 0; t < 4; ++t) {
      v16bf bfr = load_frag(brow[t], k0, half);
      acc[t] = __builtin_amdgcn_wmma_f32_16x16x32_bf16(false, af, false, bfr,
                                                       (short)0, acc[t], false, false);
    }
  }
#pragma unroll
  for (int t = 0; t < 4; ++t) {
    int col = n0 + t * 16 + idx;
    float bias = 0.0f;
    if (b0) bias += b0[col];
    if (b1) bias += b1[col];
#pragma unroll
    for (int r = 0; r < 8; ++r) {
      int row = m0 + r + half * 8;
      float v = acc[t][r] + bias;
      if (Yh) Yh[(long)row * N + col] = f2bfu(v);
      else    Yf[(long)row * N + col] = v;
    }
  }
}

// One timestep of both directions of one LSTM layer.
// gx:    [2][T][B][4H] fp32, precomputed x@Wih^T + bih + bhh (actual-time order)
// whh:   bf16, layer base; dir1 at +whh_dstride (elements); [4H][H] row-major
// h0/h1: bf16 per-dir output pointers (column offset pre-applied), row stride hstride
// cstate:[2][B][H] fp32
// Block: one 16-wide h column slice x one dir. Wave w: batch row-tile w&3,
// gates {2*(w>>2), 2*(w>>2)+1}; A-frag (h_prev) shared across both wmma.
__global__ __launch_bounds__(256) void lstm_step(
    const float* __restrict__ gx, const unsigned short* __restrict__ whh,
    long whh_dstride, unsigned short* __restrict__ h0,
    unsigned short* __restrict__ h1, int hstride,
    float* __restrict__ cstate, int H, int s) {
  int HT = H >> 4;
  int d  = blockIdx.x / HT;
  int nt = blockIdx.x % HT;
  int col0 = nt << 4;
  unsigned short* hout = (d == 0) ? h0 : h1;
  const unsigned short* w = whh + (long)d * whh_dstride;
  int t  = (d == 0) ? s : (T_STEPS - 1 - s);
  int pt = (d == 0) ? (s - 1) : (T_STEPS - s);
  const float* gxt = gx + ((long)d * T_STEPS + t) * (long)BSZ * 4 * H;

  __shared__ float g_lds[4][64][16];

  int lane = threadIdx.x & 31;
  int wave = threadIdx.x >> 5;
  int half = lane >> 4;
  int idx  = lane & 15;

  int rt = wave & 3;              // batch row tile (0..3)
  int gb = (wave >> 2) << 1;      // gate base: 0 or 2
  int m0 = rt << 4;
  v8f acc0 = {}, acc1 = {};
  if (s > 0) {                    // uniform branch; step 0 has h_prev = 0
    const unsigned short* arow  = hout + ((long)pt * BSZ + m0 + idx) * hstride;
    const unsigned short* brow0 = w + (long)((gb + 0) * H + col0 + idx) * H;
    const unsigned short* brow1 = w + (long)((gb + 1) * H + col0 + idx) * H;
    for (int k0 = 0; k0 < H; k0 += 32) {
      v16bf af  = load_frag(arow, k0, half);
      v16bf bf0 = load_frag(brow0, k0, half);
      v16bf bf1 = load_frag(brow1, k0, half);
      acc0 = __builtin_amdgcn_wmma_f32_16x16x32_bf16(false, af, false, bf0,
                                                     (short)0, acc0, false, false);
      acc1 = __builtin_amdgcn_wmma_f32_16x16x32_bf16(false, af, false, bf1,
                                                     (short)0, acc1, false, false);
    }
  }
#pragma unroll
  for (int r = 0; r < 8; ++r) {
    int row = m0 + r + half * 8;  // batch index
    g_lds[gb + 0][row][idx] = acc0[r] + gxt[(long)row * 4 * H + (gb + 0) * H + col0 + idx];
    g_lds[gb + 1][row][idx] = acc1[r] + gxt[(long)row * 4 * H + (gb + 1) * H + col0 + idx];
  }
  __syncthreads();

  // cell update: 64x16 elements over 256 threads
  for (int e = threadIdx.x; e < 64 * 16; e += blockDim.x) {
    int b = e >> 4;
    int c = e & 15;
    float iv = g_lds[0][b][c];
    float fv = g_lds[1][b][c];
    float gv = g_lds[2][b][c];
    float ov = g_lds[3][b][c];
    float* cp = cstate + ((long)d * BSZ + b) * H + col0 + c;
    float cprev = (s == 0) ? 0.0f : *cp;
    float si = 1.0f / (1.0f + __expf(-iv));
    float sf = 1.0f / (1.0f + __expf(-fv));
    float so = 1.0f / (1.0f + __expf(-ov));
    float c2 = sf * cprev + si * tanhf(gv);
    float h2 = so * tanhf(c2);
    *cp = c2;
    hout[((long)t * BSZ + b) * hstride + col0 + c] = f2bfu(h2);
  }
}

// e[t*B+b] = exp( sum_j tanh(vh[t,b,j] + sa[b,j]) * ww[j] + wb )
__global__ void att_score(const float* __restrict__ vh, const float* __restrict__ sa,
                          const float* __restrict__ ww, const float* __restrict__ wb,
                          float* __restrict__ e) {
  int tb = blockIdx.x * blockDim.x + threadIdx.x;
  if (tb >= T_STEPS * BSZ) return;
  int b = tb % BSZ;
  const float* v  = vh + (long)tb * 128;
  const float* sb = sa + (long)b * 128;
  float acc = 0.f;
  for (int j = 0; j < 128; ++j) acc += tanhf(v[j] + sb[j]) * ww[j];
  e[tb] = __expf(acc + wb[0]);
}

// c[b,j] = (1/T) * sum_t (e[t,b]/sum_t e) * h[t,b,j]   (h bf16 in, c bf16 out)
__global__ void att_context(const float* __restrict__ e,
                            const unsigned short* __restrict__ h,
                            int HH, unsigned short* __restrict__ c) {
  int b = blockIdx.x;
  __shared__ float red[256];
  red[threadIdx.x] = e[threadIdx.x * BSZ + b];   // blockDim == T_STEPS == 256
  __syncthreads();
  for (int st = 128; st > 0; st >>= 1) {
    if (threadIdx.x < st) red[threadIdx.x] += red[threadIdx.x + st];
    __syncthreads();
  }
  float denom = red[0] * (float)T_STEPS;
  for (int j = threadIdx.x; j < HH; j += blockDim.x) {
    float acc = 0.f;
    for (int t = 0; t < T_STEPS; ++t)
      acc += e[t * BSZ + b] * bf2f(h[((long)t * BSZ + b) * HH + j]);
    c[(long)b * HH + j] = f2bfu(acc / denom);
  }
}

// cross-modal beta + fusion: out[b,:] = tanh(beta0*d0 + beta1*d1)
__global__ void fuse_out(const float* __restrict__ mm0, const float* __restrict__ mm1,
                         const float* __restrict__ wg, const float* __restrict__ mmw,
                         const float* __restrict__ d0, const float* __restrict__ d1,
                         float* __restrict__ out) {
  int b = blockIdx.x;
  int j = threadIdx.x;                       // 128 threads
  __shared__ float r0[128], r1[128];
  r0[j] = tanhf(mm0[b * 128 + j] + wg[b * 128 + j]) * mmw[j];
  r1[j] = tanhf(mm1[b * 128 + j] + wg[b * 128 + j]) * mmw[j];
  __syncthreads();
  for (int st = 64; st > 0; st >>= 1) {
    if (j < st) { r0[j] += r0[j + st]; r1[j] += r1[j + st]; }
    __syncthreads();
  }
  float sc0 = r0[0], sc1 = r1[0];
  float m = fmaxf(sc0, sc1);
  float e0 = __expf(sc0 - m), e1 = __expf(sc1 - m);
  float bt0 = e0 / (e0 + e1), bt1 = e1 / (e0 + e1);
  for (int k = j; k < 512; k += 128)
    out[b * 512 + k] = tanhf(bt0 * d0[b * 512 + k] + bt1 * d1[b * 512 + k]);
}

extern "C" void kernel_launch(void* const* d_in, const int* in_sizes, int n_in,
                              void* d_out, int out_size, void* d_ws, size_t ws_size,
                              hipStream_t stream) {
  (void)in_sizes; (void)n_in; (void)out_size; (void)ws_size;
  const float* s_in   = (const float*)d_in[0];
  const float* x0     = (const float*)d_in[1];
  const float* x1     = (const float*)d_in[2];
  const float* emb0_w = (const float*)d_in[3];
  const float* emb0_b = (const float*)d_in[4];
  const float* emb1_w = (const float*)d_in[5];
  const float* emb1_b = (const float*)d_in[6];
  const float* l0_wih = (const float*)d_in[7];
  const float* l0_whh = (const float*)d_in[8];
  const float* l0_bih = (const float*)d_in[9];
  const float* l0_bhh = (const float*)d_in[10];
  const float* l1_wih = (const float*)d_in[11];
  const float* l1_whh = (const float*)d_in[12];
  const float* l1_bih = (const float*)d_in[13];
  const float* l1_bhh = (const float*)d_in[14];
  const float* atV0_w = (const float*)d_in[15];
  const float* atV0_b = (const float*)d_in[16];
  const float* atV1_w = (const float*)d_in[17];
  const float* atV1_b = (const float*)d_in[18];
  const float* atW0_w = (const float*)d_in[19];
  const float* atW0_b = (const float*)d_in[20];
  const float* atW1_w = (const float*)d_in[21];
  const float* atW1_b = (const float*)d_in[22];
  const float* atw0_w = (const float*)d_in[23];
  const float* atw0_b = (const float*)d_in[24];
  const float* atw1_w = (const float*)d_in[25];
  const float* atw1_b = (const float*)d_in[26];
  const float* lgd0_w = (const float*)d_in[27];
  const float* lgd0_b = (const float*)d_in[28];
  const float* lgd1_w = (const float*)d_in[29];
  const float* lgd1_b = (const float*)d_in[30];
  const float* mmatt0_w = (const float*)d_in[31];
  const float* mmatt0_b = (const float*)d_in[32];
  const float* mmatt1_w = (const float*)d_in[33];
  const float* mmatt1_b = (const float*)d_in[34];
  const float* qest_w = (const float*)d_in[35];
  const float* qest_b = (const float*)d_in[36];
  const float* mmw_w  = (const float*)d_in[37];
  float* out = (float*)d_out;

  const long TB = (long)T_STEPS * BSZ; // 16384
  char* base = (char*)d_ws;
  size_t off = 0;
  auto takeb = [&](size_t bytes) {
    off = (off + 255) & ~(size_t)255;
    void* p = base + off;
    off += bytes;
    return p;
  };
  // fp32 scratch
  float* gx   = (float*)takeb(2 * TB * 2048 * 4);   // reused by every LSTM layer
  float* cst  = (float*)takeb(2 * 64 * 512 * 4);    // c-state (reused)
  float* vh0  = (float*)takeb(TB * 128 * 4);
  float* vh1  = (float*)takeb(TB * 128 * 4);
  float* e0   = (float*)takeb(TB * 4);
  float* e1   = (float*)takeb(TB * 4);
  float* sa0  = (float*)takeb(64 * 128 * 4);
  float* sa1  = (float*)takeb(64 * 128 * 4);
  float* wg   = (float*)takeb(64 * 128 * 4);
  float* mm0  = (float*)takeb(64 * 128 * 4);
  float* mm1  = (float*)takeb(64 * 128 * 4);
  float* dd0  = (float*)takeb(64 * 512 * 4);
  float* dd1  = (float*)takeb(64 * 512 * 4);
  // bf16 activations
  unsigned short* x0h   = (unsigned short*)takeb(TB * 2048 * 2);
  unsigned short* x1h   = (unsigned short*)takeb(TB * 128 * 2);
  unsigned short* sh    = (unsigned short*)takeb(64 * 128 * 2);
  unsigned short* emb0h = (unsigned short*)takeb(TB * 512 * 2);
  unsigned short* emb1h = (unsigned short*)takeb(TB * 128 * 2);
  unsigned short* hs0   = (unsigned short*)takeb(2 * TB * 512 * 2);
  unsigned short* hc0   = (unsigned short*)takeb(TB * 1024 * 2);
  unsigned short* hs1   = (unsigned short*)takeb(2 * TB * 128 * 2);
  unsigned short* hc1   = (unsigned short*)takeb(TB * 256 * 2);
  unsigned short* c0h   = (unsigned short*)takeb(64 * 1024 * 2);
  unsigned short* c1h   = (unsigned short*)takeb(64 * 256 * 2);
  // bf16 weights
  unsigned short* emb0wh = (unsigned short*)takeb(512 * 2048 * 2);
  unsigned short* emb1wh = (unsigned short*)takeb(128 * 128 * 2);
  unsigned short* wih0h  = (unsigned short*)takeb(4L * 2048 * 512 * 2);
  unsigned short* whh0h  = (unsigned short*)takeb(4L * 2048 * 512 * 2);
  unsigned short* wih1h  = (unsigned short*)takeb(4L * 512 * 128 * 2);
  unsigned short* whh1h  = (unsigned short*)takeb(4L * 512 * 128 * 2);
  unsigned short* atV0wh = (unsigned short*)takeb(128 * 1024 * 2);
  unsigned short* atV1wh = (unsigned short*)takeb(128 * 256 * 2);
  unsigned short* atW0wh = (unsigned short*)takeb(128 * 128 * 2);
  unsigned short* atW1wh = (unsigned short*)takeb(128 * 128 * 2);
  unsigned short* qestwh = (unsigned short*)takeb(128 * 128 * 2);
  unsigned short* mmat0wh = (unsigned short*)takeb(128 * 1024 * 2);
  unsigned short* mmat1wh = (unsigned short*)takeb(128 * 256 * 2);
  unsigned short* lgd0wh = (unsigned short*)takeb(512 * 1024 * 2);
  unsigned short* lgd1wh = (unsigned short*)takeb(512 * 256 * 2);

  auto cast = [&](const float* src, unsigned short* dst, long n) {
    long n8 = n / 8;
    int blocks = (int)((n8 + 255) / 256);
    hipLaunchKernelGGL(cast_f32_bf16, dim3(blocks), dim3(256), 0, stream, src, dst, n8);
  };
  cast(x0, x0h, TB * 2048);
  cast(x1, x1h, TB * 128);
  cast(s_in, sh, 64 * 128);
  cast(emb0_w, emb0wh, 512 * 2048);
  cast(emb1_w, emb1wh, 128 * 128);
  cast(l0_wih, wih0h, 4L * 2048 * 512);
  cast(l0_whh, whh0h, 4L * 2048 * 512);
  cast(l1_wih, wih1h, 4L * 512 * 128);
  cast(l1_whh, whh1h, 4L * 512 * 128);
  cast(atV0_w, atV0wh, 128 * 1024);
  cast(atV1_w, atV1wh, 128 * 256);
  cast(atW0_w, atW0wh, 128 * 128);
  cast(atW1_w, atW1wh, 128 * 128);
  cast(qest_w, qestwh, 128 * 128);
  cast(mmatt0_w, mmat0wh, 128 * 1024);
  cast(mmatt1_w, mmat1wh, 128 * 256);
  cast(lgd0_w, lgd0wh, 512 * 1024);
  cast(lgd1_w, lgd1wh, 512 * 256);

  auto gemm = [&](const unsigned short* A, const unsigned short* Wt,
                  const float* b0, const float* b1,
                  float* Yf, unsigned short* Yh, int M, int N, int K) {
    long grps = (long)(M / 16) * (N / 64);
    int blocks = (int)((grps + 7) / 8);
    hipLaunchKernelGGL(wmma_gemm_bias, dim3(blocks), dim3(256), 0, stream,
                       A, Wt, b0, b1, Yf, Yh, M, N, K);
  };

  // embeddings (bf16 out: consumed only by gate GEMMs)
  gemm(x0h, emb0wh, emb0_b, nullptr, nullptr, emb0h, (int)TB, 512, 2048);
  gemm(x1h, emb1wh, emb1_b, nullptr, nullptr, emb1h, (int)TB, 128, 128);

  // ---- bilstm0 (H=512) ----
  for (int layer = 0; layer < 2; ++layer) {
    for (int d = 0; d < 2; ++d) {
      const unsigned short* A = (layer == 0) ? emb0h : (hs0 + (size_t)d * TB * 512);
      long widx = (long)d * 2 + layer;
      gemm(A, wih0h + widx * 2048 * 512, l0_bih + widx * 2048, l0_bhh + widx * 2048,
           gx + (size_t)d * TB * 2048, nullptr, (int)TB, 2048, 512);
    }
    unsigned short *h0p, *h1p; int hstride;
    if (layer == 0) { h0p = hs0; h1p = hs0 + (size_t)TB * 512; hstride = 512; }
    else            { h0p = hc0; h1p = hc0 + 512;              hstride = 1024; }
    const unsigned short* whhL = whh0h + (long)layer * 2048 * 512;
    for (int s = 0; s < T_STEPS; ++s)
      hipLaunchKernelGGL(lstm_step, dim3((512 / 16) * 2), dim3(256), 0, stream,
                         gx, whhL, (long)2 * 2048 * 512, h0p, h1p, hstride, cst, 512, s);
  }

  // ---- bilstm1 (H=128) ----
  for (int layer = 0; layer < 2; ++layer) {
    for (int d = 0; d < 2; ++d) {
      const unsigned short* A = (layer == 0) ? emb1h : (hs1 + (size_t)d * TB * 128);
      long widx = (long)d * 2 + layer;
      gemm(A, wih1h + widx * 512 * 128, l1_bih + widx * 512, l1_bhh + widx * 512,
           gx + (size_t)d * TB * 512, nullptr, (int)TB, 512, 128);
    }
    unsigned short *h0p, *h1p; int hstride;
    if (layer == 0) { h0p = hs1; h1p = hs1 + (size_t)TB * 128; hstride = 128; }
    else            { h0p = hc1; h1p = hc1 + 128;              hstride = 256; }
    const unsigned short* whhL = whh1h + (long)layer * 512 * 128;
    for (int s = 0; s < T_STEPS; ++s)
      hipLaunchKernelGGL(lstm_step, dim3((128 / 16) * 2), dim3(256), 0, stream,
                         gx, whhL, (long)2 * 512 * 128, h0p, h1p, hstride, cst, 128, s);
  }

  // ---- attention, modality 0 ----
  gemm(hc0, atV0wh, atV0_b, nullptr, vh0, nullptr, (int)TB, 128, 1024);
  gemm(sh, atW0wh, atW0_b, nullptr, sa0, nullptr, 64, 128, 128);
  hipLaunchKernelGGL(att_score, dim3((int)(TB / 256)), dim3(256), 0, stream,
                     vh0, sa0, atw0_w, atw0_b, e0);
  hipLaunchKernelGGL(att_context, dim3(64), dim3(256), 0, stream, e0, hc0, 1024, c0h);
  // ---- attention, modality 1 ----
  gemm(hc1, atV1wh, atV1_b, nullptr, vh1, nullptr, (int)TB, 128, 256);
  gemm(sh, atW1wh, atW1_b, nullptr, sa1, nullptr, 64, 128, 128);
  hipLaunchKernelGGL(att_score, dim3((int)(TB / 256)), dim3(256), 0, stream,
                     vh1, sa1, atw1_w, atw1_b, e1);
  hipLaunchKernelGGL(att_context, dim3(64), dim3(256), 0, stream, e1, hc1, 256, c1h);

  // ---- cross-modal fusion ----
  gemm(sh, qestwh, qest_b, nullptr, wg, nullptr, 64, 128, 128);
  gemm(c0h, mmat0wh, mmatt0_b, nullptr, mm0, nullptr, 64, 128, 1024);
  gemm(c1h, mmat1wh, mmatt1_b, nullptr, mm1, nullptr, 64, 128, 256);
  gemm(c0h, lgd0wh, lgd0_b, nullptr, dd0, nullptr, 64, 512, 1024);
  gemm(c1h, lgd1wh, lgd1_b, nullptr, dd1, nullptr, 64, 512, 256);
  hipLaunchKernelGGL(fuse_out, dim3(64), dim3(128), 0, stream,
                     mm0, mm1, wg, mmw_w, dd0, dd1, out);
}
